// CrossAttention_59107339928222
// MI455X (gfx1250) — compile-verified, matrix operand order checked by reading, not run on hardware
//
#include <hip/hip_runtime.h>

typedef _Float16 f16;
typedef __attribute__((ext_vector_type(4)))  f16   v4h;
typedef __attribute__((ext_vector_type(8)))  f16   v8h;
typedef __attribute__((ext_vector_type(16))) f16   v16h;
typedef __attribute__((ext_vector_type(8)))  float v8f;
typedef __attribute__((ext_vector_type(4)))  float v4f;

#define E_  512
#define H_  8
#define DH_ 64
#define LQ_ 2048
#define LK_ 4096
#define B_  2

static __device__ __forceinline__ v16h cat8(v8h lo, v8h hi) {
  return __builtin_shufflevector(lo, hi, 0,1,2,3,4,5,6,7,8,9,10,11,12,13,14,15);
}
static __device__ __forceinline__ v16h ld_frag(const f16* p) {
  v8h lo = *(const v8h*)p;
  v8h hi = *(const v8h*)(p + 16);
  return cat8(lo, hi);
}
static __device__ __forceinline__ v8f wmma32(v16h a, v16h b, v8f c) {
  // D = A(16x32 f16) * B(32x16 f16) + C(16x16 f32)
  return __builtin_amdgcn_wmma_f32_16x16x32_f16(false, a, false, b, (short)0, c, false, false);
}

// Async global->LDS copy, 16B per lane, tracked on ASYNCcnt (CDNA5 path).
static __device__ __forceinline__ void async_copy_b128(uint32_t lds_off, const f16* gaddr) {
  asm volatile("global_load_async_to_lds_b128 %0, %1, off"
               :: "v"(lds_off), "v"(gaddr) : "memory");
}
static __device__ __forceinline__ void async_wait0() {
  asm volatile("s_wait_asynccnt 0x0" ::: "memory");
}

// ---------------------------------------------------------------- convert
__global__ __launch_bounds__(256) void cvt_f32_f16(const float* __restrict__ in,
                                                   f16* __restrict__ out, int n4) {
  int i = blockIdx.x * 256 + threadIdx.x;
  if (i < n4) {
    v4f v = ((const v4f*)in)[i];
    v4h o;
    #pragma unroll
    for (int e = 0; e < 4; ++e) o[e] = (f16)v[e];
    ((v4h*)out)[i] = o;
  }
}

// ----------------------------------------------------- projection GEMM (f16 out)
// out = (X @ W^T + bias) * scale ; X:[z][Mb][512] f16, W:[512][512] f16 (row-major)
// transposed==0: out[z][row][col] ; transposed==1: out[z][col][row] (for V^T)
__global__ __launch_bounds__(256) void gemm_qkv(const f16* __restrict__ X, const f16* __restrict__ W,
                                                const float* __restrict__ bias, f16* __restrict__ out,
                                                int Mb, float scale, int transposed) {
  const int K = E_, N = E_;
  const int wave = threadIdx.x >> 5, lane = threadIdx.x & 31;
  const int l16 = lane & 15, hs = lane >> 4;
  const int z = blockIdx.z;
  X   += (size_t)z * Mb * K;
  out += (size_t)z * Mb * N;
  const int tile = blockIdx.x * 8 + wave;
  const int mt = tile >> 5;      // N/16 == 32 tiles across
  const int nt = tile & 31;
  const int row0 = mt * 16, col0 = nt * 16;
  const f16* ap0 = X + (size_t)(row0 + l16) * K + hs * 8;
  const f16* bp0 = W + (size_t)(col0 + l16) * K + hs * 8;
  v8f acc = {};
  #pragma unroll 4
  for (int k0 = 0; k0 < K; k0 += 32) {
    acc = wmma32(ld_frag(ap0 + k0), ld_frag(bp0 + k0), acc);
  }
  const float bc = bias[col0 + l16];
  if (!transposed) {
    #pragma unroll
    for (int r = 0; r < 8; ++r) {
      const int row = row0 + hs * 8 + r;
      out[(size_t)row * N + col0 + l16] = (f16)((acc[r] + bc) * scale);
    }
  } else {
    v8h o;
    #pragma unroll
    for (int r = 0; r < 8; ++r) o[r] = (f16)((acc[r] + bc) * scale);
    *(v8h*)(out + (size_t)(col0 + l16) * Mb + row0 + hs * 8) = o;  // contiguous rows
  }
}

// ----------------------------------------------------------- flash attention
// q16:[b][Lq][512] (pre-scaled by 1/sqrt(DH)), k16:[b][Lk][512], vT16:[b][512][Lk]
// ctx16:[b][Lq][512]
__global__ __launch_bounds__(256) void flash_attn(const f16* __restrict__ q16, const f16* __restrict__ k16,
                                                  const f16* __restrict__ vT16, f16* __restrict__ ctx16) {
  __shared__ f16 lK[32 * 64];       // [key 0..31][dh 0..63]
  __shared__ f16 lV[64 * 32];       // [dh 0..63][key 0..31]
  __shared__ f16 lP[8][16 * 32];    // per-wave P transpose scratch [row][key]
  const int bh = blockIdx.y, b = bh >> 3, h = bh & 7;
  const int wave = threadIdx.x >> 5, lane = threadIdx.x & 31;
  const int l16 = lane & 15, hs = lane >> 4;
  const int qrow0 = blockIdx.x * 128 + wave * 16;

  // Q fragments for this wave's 16 rows (K = 64 -> two 16x32 A fragments)
  const f16* qb = q16 + ((size_t)(b * LQ_ + qrow0 + l16)) * E_ + h * DH_ + hs * 8;
  const v16h qa0 = ld_frag(qb);
  const v16h qa1 = ld_frag(qb + 32);

  // cooperative staging indices (256 threads x 16B each per 4KB tile)
  const int t = threadIdx.x;
  const int kkey = t >> 3, kdh = (t & 7) * 8;
  const int vdh = t >> 2, vkey = (t & 3) * 8;
  const f16* kg = k16 + (size_t)b * LK_ * E_ + h * DH_ + kdh;           // + (kk0+kkey)*E_
  const f16* vg = vT16 + ((size_t)b * E_ + h * DH_ + vdh) * LK_ + vkey; // + kk0
  // LDS byte offsets (AS3 offset == low 32 bits of the generic shared pointer)
  const uint32_t lkoff = (uint32_t)(size_t)&lK[kkey * 64 + kdh];
  const uint32_t lvoff = (uint32_t)(size_t)&lV[vdh * 32 + vkey];

  float Mx[8], Ls[8];
  #pragma unroll
  for (int r = 0; r < 8; ++r) { Mx[r] = -1e30f; Ls[r] = 0.f; }
  v8f acc[4] = {};

  for (int kk0 = 0; kk0 < LK_; kk0 += 32) {
    __syncthreads();  // all waves finished reading previous chunk
    // async DMA of K/V chunk straight into LDS (ASYNCcnt tracked)
    async_copy_b128(lkoff, kg + (size_t)(kk0 + kkey) * E_);
    async_copy_b128(lvoff, vg + kk0);
    if (kk0 + 32 < LK_) {
      __builtin_prefetch(kg + (size_t)(kk0 + 32 + kkey) * E_, 0, 0);
      __builtin_prefetch(vg + kk0 + 32, 0, 0);
    }
    async_wait0();
    __syncthreads();  // all waves' async fills visible

    // S = Q * K^T for two 16-key tiles
    v8f c0 = {}, c1 = {};
    {
      const f16* kbp = &lK[l16 * 64 + hs * 8];
      c0 = wmma32(qa0, ld_frag(kbp), c0);
      c0 = wmma32(qa1, ld_frag(kbp + 32), c0);
      kbp += 16 * 64;
      c1 = wmma32(qa0, ld_frag(kbp), c1);
      c1 = wmma32(qa1, ld_frag(kbp + 32), c1);
    }

    // joint row-max across both tiles (rows are striped across 16-lane halves)
    float ml[8];
    #pragma unroll
    for (int r = 0; r < 8; ++r) ml[r] = fmaxf(c0[r], c1[r]);
    #pragma unroll
    for (int mask = 1; mask < 16; mask <<= 1) {
      #pragma unroll
      for (int r = 0; r < 8; ++r) ml[r] = fmaxf(ml[r], __shfl_xor(ml[r], mask, 32));
    }
    float al[8], rs[8];
    #pragma unroll
    for (int r = 0; r < 8; ++r) {
      const float nm = fmaxf(Mx[r], ml[r]);
      al[r] = __expf(Mx[r] - nm);
      Mx[r] = nm;
    }
    #pragma unroll
    for (int r = 0; r < 8; ++r) {
      c0[r] = __expf(c0[r] - Mx[r]);
      c1[r] = __expf(c1[r] - Mx[r]);
      rs[r] = c0[r] + c1[r];
    }
    #pragma unroll
    for (int mask = 1; mask < 16; mask <<= 1) {
      #pragma unroll
      for (int r = 0; r < 8; ++r) rs[r] += __shfl_xor(rs[r], mask, 32);
    }
    #pragma unroll
    for (int r = 0; r < 8; ++r) Ls[r] = Ls[r] * al[r] + rs[r];
    #pragma unroll
    for (int tl = 0; tl < 4; ++tl) {
      #pragma unroll
      for (int r = 0; r < 8; ++r) acc[tl][r] *= al[r];
    }

    // transpose P (C-layout -> A-layout) through per-wave LDS; both tiles give K=32
    f16* pw = lP[wave];
    #pragma unroll
    for (int r = 0; r < 8; ++r) {
      const int row = hs * 8 + r;
      pw[row * 32 + l16]      = (f16)c0[r];
      pw[row * 32 + 16 + l16] = (f16)c1[r];
    }
    const v16h pa = ld_frag(pw + l16 * 32 + hs * 8);

    // ctx += P(16x32) * V(32x64): 4 WMMAs over dh tiles, keys contiguous in lV rows
    #pragma unroll
    for (int tl = 0; tl < 4; ++tl) {
      const v16h vb = ld_frag(&lV[(tl * 16 + l16) * 32 + hs * 8]);
      acc[tl] = wmma32(pa, vb, acc[tl]);
    }
  }

  float inv[8];
  #pragma unroll
  for (int r = 0; r < 8; ++r) inv[r] = 1.f / Ls[r];
  #pragma unroll
  for (int tl = 0; tl < 4; ++tl) {
    #pragma unroll
    for (int r = 0; r < 8; ++r) {
      const size_t row = (size_t)(b * LQ_ + qrow0 + hs * 8 + r);
      ctx16[row * E_ + h * DH_ + tl * 16 + l16] = (f16)(acc[tl][r] * inv[r]);
    }
  }
}

// ---------------------------------------- output projection + bias + residual (f32)
__global__ __launch_bounds__(256) void gemm_out(const f16* __restrict__ ctx, const f16* __restrict__ W,
                                                const float* __restrict__ bias, const float* __restrict__ resid,
                                                float* __restrict__ xres) {
  const int K = E_, N = E_;
  const int wave = threadIdx.x >> 5, lane = threadIdx.x & 31;
  const int l16 = lane & 15, hs = lane >> 4;
  const int tile = blockIdx.x * 8 + wave;
  const int mt = tile >> 5, nt = tile & 31;
  const int row0 = mt * 16, col0 = nt * 16;
  const f16* ap0 = ctx + (size_t)(row0 + l16) * K + hs * 8;
  const f16* bp0 = W + (size_t)(col0 + l16) * K + hs * 8;
  v8f acc = {};
  #pragma unroll 4
  for (int k0 = 0; k0 < K; k0 += 32)
    acc = wmma32(ld_frag(ap0 + k0), ld_frag(bp0 + k0), acc);
  const float bc = bias[col0 + l16];
  #pragma unroll
  for (int r = 0; r < 8; ++r) {
    const size_t idx = (size_t)(row0 + hs * 8 + r) * N + col0 + l16;
    xres[idx] = acc[r] + bc + resid[idx];
  }
}

// ------------------------------------------------------------------ LayerNorm
__global__ __launch_bounds__(256) void ln_row(const float* __restrict__ x, const float* __restrict__ g,
                                              const float* __restrict__ be, float* __restrict__ out) {
  const int row = blockIdx.x * 8 + (threadIdx.x >> 5);
  const int lane = threadIdx.x & 31;
  const float* xr = x + (size_t)row * E_ + lane * 16;
  v4f v[4];
  float s = 0.f, s2 = 0.f;
  #pragma unroll
  for (int j = 0; j < 4; ++j) {
    v[j] = *(const v4f*)(xr + 4 * j);
    #pragma unroll
    for (int e = 0; e < 4; ++e) { s += v[j][e]; s2 += v[j][e] * v[j][e]; }
  }
  #pragma unroll
  for (int mask = 1; mask < 32; mask <<= 1) {
    s  += __shfl_xor(s, mask, 32);
    s2 += __shfl_xor(s2, mask, 32);
  }
  const float mu = s * (1.f / E_);
  const float var = fmaxf(s2 * (1.f / E_) - mu * mu, 0.f);
  const float rstd = rsqrtf(var + 1e-5f);
  float* orow = out + (size_t)row * E_ + lane * 16;
  const float* gp = g + lane * 16;
  const float* bp = be + lane * 16;
  #pragma unroll
  for (int j = 0; j < 4; ++j) {
    v4f o;
    #pragma unroll
    for (int e = 0; e < 4; ++e)
      o[e] = (v[j][e] - mu) * rstd * gp[4 * j + e] + bp[4 * j + e];
    *(v4f*)(orow + 4 * j) = o;
  }
}

// ------------------------------------------------------------------ launcher
extern "C" void kernel_launch(void* const* d_in, const int* in_sizes, int n_in,
                              void* d_out, int out_size, void* d_ws, size_t ws_size,
                              hipStream_t stream) {
  (void)in_sizes; (void)n_in; (void)out_size; (void)ws_size;
  const float* query     = (const float*)d_in[0];
  const float* key_value = (const float*)d_in[1];
  const float* Wq = (const float*)d_in[2];
  const float* bq = (const float*)d_in[3];
  const float* Wk = (const float*)d_in[4];
  const float* bk = (const float*)d_in[5];
  const float* Wv = (const float*)d_in[6];
  const float* bv = (const float*)d_in[7];
  const float* Wo = (const float*)d_in[8];
  const float* bo = (const float*)d_in[9];
  const float* gamma = (const float*)d_in[10];
  const float* beta  = (const float*)d_in[11];
  float* out = (float*)d_out;

  const size_t nQ  = (size_t)B_ * LQ_ * E_;   // 2,097,152
  const size_t nKV = (size_t)B_ * LK_ * E_;   // 4,194,304
  const size_t nW  = (size_t)E_ * E_;         //   262,144

  f16* p = (f16*)d_ws;
  f16* xq16  = p; p += nQ;
  f16* kv16  = p; p += nKV;
  f16* Wq16  = p; p += nW;
  f16* Wk16  = p; p += nW;
  f16* Wv16  = p; p += nW;
  f16* Wo16  = p; p += nW;
  f16* q16   = p; p += nQ;
  f16* k16   = p; p += nKV;
  f16* vT16  = p; p += nKV;
  f16* ctx16 = p; p += nQ;
  float* xres = (float*)p;                    // nQ floats

  auto cvt = [&](const float* src, f16* dst, size_t n) {
    const int n4 = (int)(n / 4);
    cvt_f32_f16<<<dim3((n4 + 255) / 256), dim3(256), 0, stream>>>(src, dst, n4);
  };
  cvt(query, xq16, nQ);
  cvt(key_value, kv16, nKV);
  cvt(Wq, Wq16, nW);
  cvt(Wk, Wk16, nW);
  cvt(Wv, Wv16, nW);
  cvt(Wo, Wo16, nW);

  // Q = (x @ Wq^T + bq) * 1/sqrt(DH)  (batch folded into M)
  gemm_qkv<<<dim3((4096 / 16) * (512 / 16) / 8, 1, 1), 256, 0, stream>>>(xq16, Wq16, bq, q16, 4096, 0.125f, 0);
  // K = kv @ Wk^T + bk
  gemm_qkv<<<dim3((8192 / 16) * (512 / 16) / 8, 1, 1), 256, 0, stream>>>(kv16, Wk16, bk, k16, 8192, 1.0f, 0);
  // V^T = (kv @ Wv^T + bv)^T per batch (grid.z = B)
  gemm_qkv<<<dim3((4096 / 16) * (512 / 16) / 8, 1, 2), 256, 0, stream>>>(kv16, Wv16, bv, vT16, 4096, 1.0f, 1);

  flash_attn<<<dim3(LQ_ / 128, B_ * H_, 1), 256, 0, stream>>>(q16, k16, vT16, ctx16);

  gemm_out<<<dim3((4096 / 16) * (512 / 16) / 8, 1, 1), 256, 0, stream>>>(ctx16, Wo16, bo, query, xres);

  ln_row<<<dim3((B_ * LQ_) / 8, 1, 1), 256, 0, stream>>>(xres, gamma, beta, out);
}
